// ScaledDotProductAttention_85186381349423
// MI455X (gfx1250) — matrix operand order repaired and checked by reading, not compiled, not run on hardware
//
#include <hip/hip_runtime.h>

typedef __attribute__((ext_vector_type(16))) _Float16 v16h;
typedef __attribute__((ext_vector_type(4)))  _Float16 v4h;
typedef __attribute__((ext_vector_type(8)))  float    v8f;

#define S_LEN   2048
#define D_DIM   64
#define HD      4        // heads
#define BATCH   16
#define ROWS    16       // query rows per workgroup
#define NWAVES  8
#define STRIP   256      // S_LEN / NWAVES
#define PSTR    2112     // padded LDS row stride in halfs: (2112*2/4) % 64 == 32

__global__ __launch_bounds__(256)
void sdpa_wmma_kernel(const float* __restrict__ Q,
                      const float* __restrict__ K,
                      const float* __restrict__ V,
                      const int*   __restrict__ mask,
                      float* __restrict__ out,
                      float* __restrict__ attn)
{
    // LDS: unnormalized probabilities (f16) for the 16-row tile + reduction scratch
    __shared__ _Float16 Pl[ROWS * PSTR];         // ~66 KB (padded rows)
    __shared__ float    Ol[ROWS * D_DIM];        // 4 KB  partial output accumulator
    __shared__ float    red[NWAVES * ROWS];      // cross-wave max reduction
    __shared__ float    finmax[ROWS];
    __shared__ float    finsum[ROWS];

    const int tid  = threadIdx.x;
    const int wave = tid >> 5;
    const int lane = tid & 31;
    const int hl   = lane & 15;          // M (A) or N (B/C) index
    const bool hi  = lane >= 16;
    const int rowbase = hi ? 8 : 0;      // C/D: VGPR i -> row i (lanes 0-15) / i+8 (16-31)

    const int blk = blockIdx.x;
    const int rt  = blk & 127;           // 128 row tiles per (b,h)
    const int bh  = blk >> 7;            // 0..63
    const int r0  = rt * ROWS;
    const int b   = bh >> 2;

    const size_t baseQK = (size_t)bh * S_LEN * D_DIM;
    const float* Qb = Q + baseQK;
    const float* Kb = K + baseQK;
    const float* Vb = V + baseQK;
    const int*   mb = mask + (size_t)b * S_LEN;

    // zero output accumulator (visible after first barrier)
    for (int i = tid; i < ROWS * D_DIM; i += 256) Ol[i] = 0.0f;

    // ---- Load Q A-fragments (2 chunks of K=32 cover D=64) -------------------
    // 16-bit A 16x32 layout: lane<16 -> K {0..7,16..23}; lane>=16 -> K {8..15,24..31}
    v16h aq[2];
    {
        const int kb0 = hi ? 8 : 0;
        const float* qrow = Qb + (size_t)(r0 + hl) * D_DIM;
        #pragma unroll
        for (int c = 0; c < 2; ++c) {
            const float* qp = qrow + 32 * c + kb0;
            float4 a0 = ((const float4*)qp)[0];
            float4 a1 = ((const float4*)qp)[1];
            float4 a2 = ((const float4*)(qp + 16))[0];
            float4 a3 = ((const float4*)(qp + 16))[1];
            aq[c][0]=(_Float16)a0.x; aq[c][1]=(_Float16)a0.y; aq[c][2]=(_Float16)a0.z; aq[c][3]=(_Float16)a0.w;
            aq[c][4]=(_Float16)a1.x; aq[c][5]=(_Float16)a1.y; aq[c][6]=(_Float16)a1.z; aq[c][7]=(_Float16)a1.w;
            aq[c][8]=(_Float16)a2.x; aq[c][9]=(_Float16)a2.y; aq[c][10]=(_Float16)a2.z; aq[c][11]=(_Float16)a2.w;
            aq[c][12]=(_Float16)a3.x; aq[c][13]=(_Float16)a3.y; aq[c][14]=(_Float16)a3.z; aq[c][15]=(_Float16)a3.w;
        }
    }

    // ---- Phase 1: scores = (Q K^T)/16 with mask; track row max; stash f16 ---
    const int koff = hi ? 8 : 0;         // B 32x16: lanes<16 K={0..7,16..23}, lanes>=16 K={8..15,24..31}
    float mrow[8];
    #pragma unroll
    for (int i = 0; i < 8; ++i) mrow[i] = -3.0e38f;

    for (int t = 0; t < 16; ++t) {
        const int c0 = wave * STRIP + t * 16;
        v16h bk[2];
        const float* krow = Kb + (size_t)(c0 + hl) * D_DIM + koff;
        #pragma unroll
        for (int c = 0; c < 2; ++c) {
            const float* kp = krow + 32 * c;
            float4 x0 = ((const float4*)kp)[0];
            float4 x1 = ((const float4*)kp)[1];
            float4 x2 = ((const float4*)(kp + 16))[0];
            float4 x3 = ((const float4*)(kp + 16))[1];
            bk[c][0]=(_Float16)x0.x; bk[c][1]=(_Float16)x0.y; bk[c][2]=(_Float16)x0.z; bk[c][3]=(_Float16)x0.w;
            bk[c][4]=(_Float16)x1.x; bk[c][5]=(_Float16)x1.y; bk[c][6]=(_Float16)x1.z; bk[c][7]=(_Float16)x1.w;
            bk[c][8]=(_Float16)x2.x; bk[c][9]=(_Float16)x2.y; bk[c][10]=(_Float16)x2.z; bk[c][11]=(_Float16)x2.w;
            bk[c][12]=(_Float16)x3.x; bk[c][13]=(_Float16)x3.y; bk[c][14]=(_Float16)x3.z; bk[c][15]=(_Float16)x3.w;
        }
        v8f acc = {};
        acc = __builtin_amdgcn_wmma_f32_16x16x32_f16(false, aq[0], false, bk[0], (short)0, acc, false, false);
        acc = __builtin_amdgcn_wmma_f32_16x16x32_f16(false, aq[1], false, bk[1], (short)0, acc, false, false);

        const int mv = mb[c0 + hl];
        #pragma unroll
        for (int i = 0; i < 8; ++i) {
            float s = acc[i] * 0.0625f;             // 1/sqrt(256)
            s = (mv == 0) ? -1.0e9f : s;
            mrow[i] = fmaxf(mrow[i], s);
            Pl[(i + rowbase) * PSTR + c0 + hl] = (_Float16)s;
        }
    }
    // in-wave row-max (reduce across the 16 lanes holding one row)
    #pragma unroll
    for (int i = 0; i < 8; ++i) {
        float v = mrow[i];
        v = fmaxf(v, __shfl_xor(v, 1, 32));
        v = fmaxf(v, __shfl_xor(v, 2, 32));
        v = fmaxf(v, __shfl_xor(v, 4, 32));
        v = fmaxf(v, __shfl_xor(v, 8, 32));
        mrow[i] = v;
    }
    if (hl == 0) {
        #pragma unroll
        for (int i = 0; i < 8; ++i) red[wave * ROWS + i + rowbase] = mrow[i];
    }
    __syncthreads();
    if (tid < ROWS) {
        float v = red[tid];
        for (int w = 1; w < NWAVES; ++w) v = fmaxf(v, red[w * ROWS + tid]);
        finmax[tid] = v;
    }
    __syncthreads();

    // ---- Phase 2: row-major vectorized exp/sum; rewrite LDS with unnorm p ---
    // thread t owns row (t>>4); 16 threads per row split 2048 cols; each half-wave
    // is one row -> 16-lane shfl reduction yields the COMPLETE row sum.
    const int row2 = tid >> 4;
    const int seg  = tid & 15;
    {
        const float fm2 = finmax[row2];
        float ssum = 0.0f;
        _Float16* prow = &Pl[row2 * PSTR];
        for (int iter = 0; iter < 32; ++iter) {
            const int col = seg * 4 + iter * 64;
            v4h* pp = (v4h*)(prow + col);
            v4h s4 = *pp;
            float p0 = __expf((float)s4[0] - fm2);
            float p1 = __expf((float)s4[1] - fm2);
            float p2 = __expf((float)s4[2] - fm2);
            float p3 = __expf((float)s4[3] - fm2);
            ssum += (p0 + p1) + (p2 + p3);
            s4[0] = (_Float16)p0; s4[1] = (_Float16)p1;
            s4[2] = (_Float16)p2; s4[3] = (_Float16)p3;
            *pp = s4;
        }
        ssum += __shfl_xor(ssum, 1, 32);
        ssum += __shfl_xor(ssum, 2, 32);
        ssum += __shfl_xor(ssum, 4, 32);
        ssum += __shfl_xor(ssum, 8, 32);
        if (seg == 0) finsum[row2] = ssum;
    }
    __syncthreads();

    // ---- Phase 3: normalized attn_dist stores, float4/lane (256B/half-wave) -
    {
        const float inv2 = 1.0f / finsum[row2];
        const _Float16* prow = &Pl[row2 * PSTR];
        float* arow = attn + ((size_t)bh * S_LEN + r0 + row2) * S_LEN;
        for (int iter = 0; iter < 32; ++iter) {
            const int col = seg * 4 + iter * 64;
            v4h p4 = *(const v4h*)(prow + col);
            float4 o;
            o.x = (float)p4[0] * inv2;
            o.y = (float)p4[1] * inv2;
            o.z = (float)p4[2] * inv2;
            o.w = (float)p4[3] * inv2;
            *(float4*)(arow + col) = o;
        }
    }

    // ---- Phase 4: O = P_unnorm * V over this wave's K-range; reduce in LDS --
    v8f oacc[4] = { {}, {}, {}, {} };    // four 16-wide column tiles of D=64
    const int kb0A = hi ? 8 : 0;
    for (int kb = 0; kb < 8; ++kb) {
        const int kbase = wave * STRIP + kb * 32;
        v16h ap;
        const _Float16* prow = &Pl[hl * PSTR + kbase];
        #pragma unroll
        for (int j = 0; j < 8; ++j) ap[j]     = prow[kb0A + j];
        #pragma unroll
        for (int j = 0; j < 8; ++j) ap[8 + j] = prow[16 + kb0A + j];

        #pragma unroll
        for (int f = 0; f < 4; ++f) {
            v16h bv;
            const float* vcol = Vb + (size_t)(kbase + koff) * D_DIM + f * 16 + hl;
            #pragma unroll
            for (int j = 0; j < 16; ++j) bv[j] = (_Float16)vcol[(j < 8 ? j : j + 8) * D_DIM];
            oacc[f] = __builtin_amdgcn_wmma_f32_16x16x32_f16(false, ap, false, bv, (short)0, oacc[f], false, false);
        }
    }
    #pragma unroll
    for (int f = 0; f < 4; ++f)
        #pragma unroll
        for (int i = 0; i < 8; ++i)
            atomicAdd(&Ol[(i + rowbase) * D_DIM + f * 16 + hl], oacc[f][i]);
    __syncthreads();

    // ---- Phase 5: scale by 1/rowsum and store output (float4 per thread) ----
    {
        const int idx = tid * 4;                 // 1024 floats / 256 threads
        const int row = idx >> 6;
        const int d0  = idx & 63;
        const float inv = 1.0f / finsum[row];
        float4 o;
        o.x = Ol[idx + 0] * inv;
        o.y = Ol[idx + 1] * inv;
        o.z = Ol[idx + 2] * inv;
        o.w = Ol[idx + 3] * inv;
        float* op = out + ((size_t)bh * S_LEN + r0 + row) * D_DIM + d0;
        *(float4*)op = o;
    }
}

extern "C" void kernel_launch(void* const* d_in, const int* in_sizes, int n_in,
                              void* d_out, int out_size, void* d_ws, size_t ws_size,
                              hipStream_t stream) {
    (void)in_sizes; (void)n_in; (void)out_size; (void)d_ws; (void)ws_size;
    const float* Q  = (const float*)d_in[0];
    const float* K  = (const float*)d_in[1];
    const float* V  = (const float*)d_in[2];
    const int* mask = (const int*)d_in[3];

    float* out  = (float*)d_out;
    float* attn = out + (size_t)BATCH * HD * S_LEN * D_DIM;   // outputs concatenated

    const int grid = BATCH * HD * (S_LEN / ROWS);             // 16*4*128 = 8192
    sdpa_wmma_kernel<<<grid, 256, 0, stream>>>(Q, K, V, mask, out, attn);
}
